// Head_575525617925
// MI455X (gfx1250) — matrix operand order repaired and checked by reading, not compiled, not run on hardware
//
#include <hip/hip_runtime.h>

#define BB 4
#define TT 2048
#define CC 1024
#define HH 64

typedef __bf16 bf16;
typedef __attribute__((ext_vector_type(16))) bf16 v16bf;
typedef __attribute__((ext_vector_type(8)))  bf16 v8bf;
typedef __attribute__((ext_vector_type(4)))  bf16 v4bf;
typedef __attribute__((ext_vector_type(8)))  float v8f;
typedef __attribute__((ext_vector_type(4)))  float f32x4;
typedef __attribute__((ext_vector_type(4)))  unsigned int u32x4;
typedef __attribute__((ext_vector_type(8)))  int i32x8;
typedef __attribute__((ext_vector_type(4)))  int i32x4;

static __device__ __forceinline__ v8f wmma_bf16(v16bf a, v16bf b, v8f c) {
  // (neg_a, A, neg_b, B, c_mod, C, reuse_a, reuse_b)
  return __builtin_amdgcn_wmma_f32_16x16x32_bf16(false, a, false, b, (short)0, c,
                                                 false, false);
}

// A-operand (16x32 bf16): M = lane&15, element e -> K = e + 16*(lane>>4)
// => 16 contiguous values per lane: [16h .. 16h+15]
static __device__ __forceinline__ v16bf afrag_bf16(const bf16* row, int h) {
  v8bf lo = *(const v8bf*)(row + 16 * h);
  v8bf hi = *(const v8bf*)(row + 16 * h + 8);
  v16bf r;
#pragma unroll
  for (int i = 0; i < 8; ++i) { r[i] = lo[i]; r[8 + i] = hi[i]; }
  return r;
}

// B-operand (32x16 bf16): N = lane&15, element e -> K = e + 16*(lane>>4)
// => 16 contiguous values per lane
static __device__ __forceinline__ v16bf bfrag_bf16(const bf16* col, int h) {
  return *(const v16bf*)(col + 16 * h);
}

// ---------------------------------------------------------------------------
// Stage 0: one-shot fp32 -> bf16 conversion of x and the three weights.
// Removes all per-tile conversion VALU from the GEMM loops and halves the
// HBM bytes the later kernels pull.
// ---------------------------------------------------------------------------
__global__ __launch_bounds__(256) void cvt_kernel(
    const float* __restrict__ x, const float* __restrict__ Wk,
    const float* __restrict__ Wq, const float* __restrict__ Wv,
    bf16* __restrict__ xb, bf16* __restrict__ wqb, bf16* __restrict__ wkb,
    bf16* __restrict__ wvb) {
  const size_t i = (size_t)blockIdx.x * blockDim.x + threadIdx.x;  // quad idx
  const size_t nx = (size_t)BB * TT * CC / 4;
  if (i < nx) {
    f32x4 v = *(const f32x4*)(x + i * 4);
    v4bf o;
#pragma unroll
    for (int j = 0; j < 4; ++j) o[j] = (bf16)v[j];
    *(v4bf*)(xb + i * 4) = o;
  }
  const size_t nw = (size_t)HH * CC / 4;
  if (i < nw) {
    f32x4 a = *(const f32x4*)(Wq + i * 4);
    f32x4 b = *(const f32x4*)(Wk + i * 4);
    f32x4 c = *(const f32x4*)(Wv + i * 4);
    v4bf oa, ob, oc;
#pragma unroll
    for (int j = 0; j < 4; ++j) {
      oa[j] = (bf16)a[j]; ob[j] = (bf16)b[j]; oc[j] = (bf16)c[j];
    }
    *(v4bf*)(wqb + i * 4) = oa;
    *(v4bf*)(wkb + i * 4) = ob;
    *(v4bf*)(wvb + i * 4) = oc;
  }
}

// ---------------------------------------------------------------------------
// Stage 1: Q/K/V projections.  xb:[B*T, C] bf16 staged into LDS by the
// Tensor Data Mover (one 16x1024 tile per block); 3 waves {Q,K,V} consume it.
// Q,K stored bf16 row-major [B*T][64]; V stored transposed bf16 [B][64][T].
// ---------------------------------------------------------------------------
__global__ __launch_bounds__(96) void proj_kernel(
    const bf16* __restrict__ xb, const bf16* __restrict__ wqb,
    const bf16* __restrict__ wkb, const bf16* __restrict__ wvb,
    bf16* __restrict__ qo, bf16* __restrict__ ko, bf16* __restrict__ vt) {
  __shared__ bf16 xtile[16 * CC];  // 32 KB of the 320 KB/WGP LDS

  const int lane = threadIdx.x & 31;
  const int wave = threadIdx.x >> 5;  // 0:Q 1:K 2:V
  const int h = lane >> 4, lc = lane & 15;
  const int row0 = blockIdx.x * 16;  // token tile

  if (wave == 0) {
    // --- TDM: DMA the 16x1024 bf16 tile (32 KB) global -> LDS -------------
    const unsigned long long ga =
        (unsigned long long)(uintptr_t)(xb + (size_t)row0 * CC);
    const unsigned ldsoff = (unsigned)(uintptr_t)(void*)&xtile[0];
    // D# group 0: count=1, lds_addr, global_addr[56:0], type=2 ("image")
    u32x4 g0;
    g0[0] = 1u;                                   // count=1, user mode
    g0[1] = ldsoff;                               // LDS byte address
    g0[2] = (unsigned)ga;                         // global_addr[31:0]
    g0[3] = (unsigned)((ga >> 32) & 0x01FFFFFFu)  // global_addr[56:32]
            | (2u << 30);                         // type=2
    // D# group 1: data_size=2B, tensor 1024x8192, tile 1024x16, stride0=1024
    i32x8 g1;
    g1[0] = 0x00010000;            // workgroup_mask=0, data_size=1 (2 bytes)
    g1[1] = (int)(1024u << 16);    // tensor_dim0[15:0]=1024
    g1[2] = (int)(8192u << 16);    // tensor_dim0[31:16]=0, tensor_dim1[15:0]=8192
    g1[3] = (int)(1024u << 16);    // tensor_dim1[31:16]=0, tile_dim0=1024
    g1[4] = 16;                    // tile_dim1=16, tile_dim2=0
    g1[5] = 1024;                  // tensor_dim0_stride[31:0]=1024
    g1[6] = 0;                     // stride0[47:32]=0, stride1[15:0]=0
    g1[7] = 0;                     // stride1[47:16]=0
    i32x4 gz = {0, 0, 0, 0};       // groups 2/3 unused (2-D tensor)
    i32x8 gz8 = {0, 0, 0, 0, 0, 0, 0, 0};
    __builtin_amdgcn_tensor_load_to_lds(g0, g1, gz, gz, gz8, 0);
    __builtin_amdgcn_s_wait_tensorcnt((short)0);
  }
  __syncthreads();

  const bf16* W = (wave == 0) ? wqb : (wave == 1) ? wkb : wvb;
  const bf16* xrow = &xtile[(size_t)lc * CC];

  const v8f zf = {0.f, 0.f, 0.f, 0.f, 0.f, 0.f, 0.f, 0.f};
  v8f acc[4] = {zf, zf, zf, zf};

  for (int kk = 0; kk < CC; kk += 32) {
    v16bf a = afrag_bf16(xrow + kk, h);  // ds_load from staged tile
#pragma unroll
    for (int nt = 0; nt < 4; ++nt) {
      // B[c, n] = W[n, c];  column n = nt*16 + lc
      v16bf b = bfrag_bf16(W + (size_t)(nt * 16 + lc) * CC + kk, h);
      acc[nt] = wmma_bf16(a, b, acc[nt]);
    }
  }

  // D layout: lane -> col N = lane&15, vgpr v -> row M = v + 8*(lane>>4)
  if (wave < 2) {
    bf16* o = wave ? ko : qo;
#pragma unroll
    for (int nt = 0; nt < 4; ++nt)
#pragma unroll
      for (int v = 0; v < 8; ++v)
        o[(size_t)(row0 + v + 8 * h) * HH + nt * 16 + lc] = (bf16)acc[nt][v];
  } else {
    const int b = row0 >> 11;  // row0 / T
    const int tloc = row0 & (TT - 1);
#pragma unroll
    for (int nt = 0; nt < 4; ++nt)
#pragma unroll
      for (int v = 0; v < 8; ++v)
        vt[((size_t)b * HH + nt * 16 + lc) * TT + tloc + v + 8 * h] =
            (bf16)acc[nt][v];
  }
}

// ---------------------------------------------------------------------------
// Stage 2: flash attention.  One wave per 16-query tile.
// S^T = K Q^T  (keys = rows across VGPRs, queries = cols across lanes), so
// softmax reductions are per-lane + one shfl_xor(16).  O^T = V^T P^T.
// ---------------------------------------------------------------------------
#define AT_WAVES 8
__global__ __launch_bounds__(32 * AT_WAVES) void attn_kernel(
    const bf16* __restrict__ qws, const bf16* __restrict__ kws,
    const bf16* __restrict__ vws, float* __restrict__ out) {
  const int lane = threadIdx.x & 31;
  const int wave = threadIdx.x >> 5;
  const int h = lane >> 4, lc = lane & 15;
  const int tile = blockIdx.x * AT_WAVES + wave;
  const int b = tile / (TT / 16);
  const int qbase = (tile % (TT / 16)) * 16;

  const bf16* Q = qws + (size_t)b * TT * HH;
  const bf16* K = kws + (size_t)b * TT * HH;
  const bf16* Vt = vws + (size_t)b * HH * TT;
  const float scale = 0.03125f;  // C^-0.5 = 1/32

  // Q^T B-operand fragments for head-dim steps 0..31 and 32..63
  v16bf qf0, qf1;
  {
    const bf16* qrow = Q + (size_t)(qbase + lc) * HH;
    qf0 = bfrag_bf16(qrow, h);
    qf1 = bfrag_bf16(qrow + 32, h);
  }

  const v8f zf = {0.f, 0.f, 0.f, 0.f, 0.f, 0.f, 0.f, 0.f};
  v8f o[4] = {zf, zf, zf, zf};  // O^T accumulators, head blocks of 16
  float m_run = -1e30f, l_run = 0.f;
  const int qg = qbase + lc;          // this lane's query index
  const int nkv = (qbase + 47) >> 5;  // causal: keys 0 .. qbase+15

  for (int kb = 0; kb < nkv; ++kb) {
    const int kv0 = kb << 5;
    if (kv0 + 32 < TT)  // gfx1250 global_prefetch_b8 for next K block
      __builtin_prefetch(K + (size_t)(kv0 + 32 + lc) * HH, 0, 1);

    // S^T (32 keys x 16 queries) as two 16x16 D tiles
    const bf16* kr0 = K + (size_t)(kv0 + lc) * HH;
    const bf16* kr1 = kr0 + 16 * HH;
    v8f s0 = zf, s1 = zf;
    s0 = wmma_bf16(afrag_bf16(kr0, h), qf0, s0);
    s1 = wmma_bf16(afrag_bf16(kr1, h), qf0, s1);
    s0 = wmma_bf16(afrag_bf16(kr0 + 32, h), qf1, s0);
    s1 = wmma_bf16(afrag_bf16(kr1 + 32, h), qf1, s1);

    // scale + causal mask + running max (key row = kv0 + 16t + v + 8h)
    float mx = -1e30f;
#pragma unroll
    for (int v = 0; v < 8; ++v) {
      const int k0 = kv0 + v + 8 * h, k1 = k0 + 16;
      float f0 = (k0 <= qg) ? s0[v] * scale : -1e30f;
      float f1 = (k1 <= qg) ? s1[v] * scale : -1e30f;
      s0[v] = f0; s1[v] = f1;
      mx = fmaxf(mx, fmaxf(f0, f1));
    }
    mx = fmaxf(mx, __shfl_xor(mx, 16, 32));
    const float m_new = fmaxf(m_run, mx);
    const float alpha = __expf(m_run - m_new);

    float rs = 0.f;
#pragma unroll
    for (int v = 0; v < 8; ++v) {
      float p0 = __expf(s0[v] - m_new);
      float p1 = __expf(s1[v] - m_new);
      s0[v] = p0; s1[v] = p1;
      rs += p0 + p1;
    }
    rs += __shfl_xor(rs, 16, 32);
    l_run = l_run * alpha + rs;
    m_run = m_new;

#pragma unroll
    for (int t = 0; t < 4; ++t)
#pragma unroll
      for (int v = 0; v < 8; ++v) o[t][v] *= alpha;

    // Repack P^T (D layout) into B-operand layout: element e needs
    // tile (lane>>4), row e, col lc -> lane-local except half-wave exchange.
    v8f x0, x1;
#pragma unroll
    for (int v = 0; v < 8; ++v) {
      x0[v] = __shfl_xor(s0[v], 16, 32);
      x1[v] = __shfl_xor(s1[v], 16, 32);
    }
    v16bf pb;
#pragma unroll
    for (int e = 0; e < 8; ++e) pb[e] = (bf16)(h ? x1[e] : s0[e]);
#pragma unroll
    for (int e = 0; e < 8; ++e) pb[8 + e] = (bf16)(h ? s1[e] : x0[e]);

    // O^T += V^T P^T ; V^T is contiguous along keys thanks to transposed store
#pragma unroll
    for (int t = 0; t < 4; ++t) {
      const bf16* vr = Vt + (size_t)(t * 16 + lc) * TT + kv0;
      o[t] = wmma_bf16(afrag_bf16(vr, h), pb, o[t]);
    }
  }

  const float inv = 1.f / l_run;
  float* orow = out + ((size_t)b * TT + qbase + lc) * HH;
#pragma unroll
  for (int t = 0; t < 4; ++t)
#pragma unroll
    for (int v = 0; v < 8; ++v)
      orow[t * 16 + v + 8 * h] = o[t][v] * inv;
}

extern "C" void kernel_launch(void* const* d_in, const int* in_sizes, int n_in,
                              void* d_out, int out_size, void* d_ws,
                              size_t ws_size, hipStream_t stream) {
  (void)in_sizes; (void)n_in; (void)out_size; (void)ws_size;
  const float* x  = (const float*)d_in[0];
  const float* Wk = (const float*)d_in[1];
  const float* Wq = (const float*)d_in[2];
  const float* Wv = (const float*)d_in[3];
  float* out = (float*)d_out;

  bf16* xb  = (bf16*)d_ws;                        // [B*T][C] bf16, 16 MB
  bf16* wqb = xb + (size_t)BB * TT * CC;          // [64][C] bf16, 128 KB
  bf16* wkb = wqb + (size_t)HH * CC;
  bf16* wvb = wkb + (size_t)HH * CC;
  bf16* qws = wvb + (size_t)HH * CC;              // [B*T][64] bf16, 1 MB
  bf16* kws = qws + (size_t)BB * TT * HH;
  bf16* vws = kws + (size_t)BB * TT * HH;         // [B][64][T] bf16, 1 MB

  const size_t nquads = (size_t)BB * TT * CC / 4;
  cvt_kernel<<<(int)((nquads + 255) / 256), 256, 0, stream>>>(
      x, Wk, Wq, Wv, xb, wqb, wkb, wvb);
  proj_kernel<<<(BB * TT) / 16, 96, 0, stream>>>(xb, wqb, wkb, wvb, qws, kws,
                                                 vws);
  attn_kernel<<<(BB * (TT / 16)) / AT_WAVES, 32 * AT_WAVES, 0, stream>>>(
      qws, kws, vws, out);
}
